// RTMDetPipeline_25769803776046
// MI455X (gfx1250) — matrix operand order repaired and checked by reading, not compiled, not run on hardware
//
#include <hip/hip_runtime.h>
#include <hip/hip_bf16.h>
#include <math.h>

typedef __attribute__((ext_vector_type(16))) _Float16 v16h;
typedef __attribute__((ext_vector_type(8)))  float    v8f;

#define NCLS 80
#define NA 8400
#define KTOP 100
#define PRE_THR 0.3f
#define IOU_THR 0.3f
#define NEGV (-1000000000.0f)
#define PSTRIDE 253
#define HW 6400

__device__ __forceinline__ int clampi(int v, int lo, int hi) {
    return v < lo ? lo : (v > hi ? hi : v);
}

// ---------------- Kernel 1: per-anchor class argmax + sigmoid + pre-threshold ----
__global__ void k_scores(const float* __restrict__ preds,
                         float* __restrict__ scores, int* __restrict__ labels) {
    int i = blockIdx.x * blockDim.x + threadIdx.x;
    if (i >= 2 * NA) return;
    const float* p = preds + (size_t)i * PSTRIDE;
    float best = p[0]; int arg = 0;
    for (int c = 1; c < NCLS; ++c) {
        float v = p[c];
        if (v > best) { best = v; arg = c; }
    }
    float s = 1.0f / (1.0f + __expf(-best));   // sigmoid monotonic -> max/argmax preserved
    scores[i] = (s > PRE_THR) ? s : NEGV;
    labels[i] = arg;
}

// ---------------- Kernel 2: top-K=100 per batch (scores staged in LDS) ----------
__global__ void k_topk(const float* __restrict__ scores,
                       float* __restrict__ tops, int* __restrict__ topidx) {
    __shared__ float sv[NA];
    __shared__ float rv[256];
    __shared__ int   ri[256];
    int b = blockIdx.x, t = threadIdx.x;
    const float* sb = scores + b * NA;
    for (int i = t; i < NA; i += 256) sv[i] = sb[i];
    __syncthreads();
    for (int k = 0; k < KTOP; ++k) {
        float bv = -INFINITY; int bi = NA;
        for (int i = t; i < NA; i += 256) {
            float v = sv[i];
            if (v > bv || (v == bv && i < bi)) { bv = v; bi = i; }
        }
        rv[t] = bv; ri[t] = bi;
        __syncthreads();
        for (int off = 128; off > 0; off >>= 1) {
            if (t < off) {
                float ov = rv[t + off]; int oi = ri[t + off];
                if (ov > rv[t] || (ov == rv[t] && oi < ri[t])) { rv[t] = ov; ri[t] = oi; }
            }
            __syncthreads();
        }
        if (t == 0) {
            tops[b * KTOP + k]   = rv[0];
            topidx[b * KTOP + k] = ri[0];
            sv[ri[0]] = -INFINITY;
        }
        __syncthreads();
    }
}

// ---------------- Kernel 3: box decode + class-aware NMS ------------------------
__global__ void k_nms(const float* __restrict__ preds, const float* __restrict__ priors,
                      const int* __restrict__ labels, const float* __restrict__ tops,
                      const int* __restrict__ topidx,
                      float* __restrict__ out, int* __restrict__ finalf) {
    __shared__ float bo[KTOP][4];
    __shared__ float labf[KTOP];
    __shared__ int   vm[KTOP];
    __shared__ int   supp[KTOP];
    __shared__ float red[128];
    int b = blockIdx.x, t = threadIdx.x;
    float ts = NEGV;
    float bx0 = 0.f, bx1 = 0.f, bx2 = 0.f, bx3 = 0.f;
    if (t < KTOP) {
        int idx = topidx[b * KTOP + t];
        ts = tops[b * KTOP + t];
        const float* pr = preds + (size_t)(b * NA + idx) * PSTRIDE;
        float px = priors[idx * 4 + 0], py = priors[idx * 4 + 1];
        bx0 = px - pr[NCLS + 0]; bx1 = py - pr[NCLS + 1];
        bx2 = px + pr[NCLS + 2]; bx3 = py + pr[NCLS + 3];
        int lab = labels[b * NA + idx];
        vm[t]   = (ts > PRE_THR) ? 1 : 0;
        labf[t] = (float)lab;
        supp[t] = 0;
        bo[t][0] = bx0; bo[t][1] = bx1; bo[t][2] = bx2; bo[t][3] = bx3;
        // outputs: boxes @0, sc @800, lab @1000
        out[b * KTOP * 4 + t * 4 + 0] = bx0;
        out[b * KTOP * 4 + t * 4 + 1] = bx1;
        out[b * KTOP * 4 + t * 4 + 2] = bx2;
        out[b * KTOP * 4 + t * 4 + 3] = bx3;
        out[2 * KTOP * 4 + 2 * KTOP + b * KTOP + t] = (float)lab;
    }
    // maxc = max over where(vmask, boxes, 0)
    float mc = -INFINITY;
    if (t < KTOP) {
        mc = vm[t] ? fmaxf(fmaxf(bx0, bx1), fmaxf(bx2, bx3)) : 0.0f;
    }
    red[t] = mc;
    __syncthreads();
    for (int off = 64; off > 0; off >>= 1) {
        if (t < off) red[t] = fmaxf(red[t], red[t + off]);
        __syncthreads();
    }
    float maxc = red[0];
    __syncthreads();
    if (t < KTOP) {
        float o = labf[t] * (maxc + 1.0f);
        bo[t][0] += o; bo[t][1] += o; bo[t][2] += o; bo[t][3] += o;
    }
    __syncthreads();
    for (int i = 0; i < KTOP; ++i) {
        int alive = (!supp[i]) && vm[i];
        if (alive && t > i && t < KTOP) {
            float ax0 = bo[i][0], ay0 = bo[i][1], ax1 = bo[i][2], ay1 = bo[i][3];
            float cx0 = bo[t][0], cy0 = bo[t][1], cx1 = bo[t][2], cy1 = bo[t][3];
            float areaA = (ax1 - ax0) * (ay1 - ay0);
            float areaB = (cx1 - cx0) * (cy1 - cy0);
            float lx = fmaxf(ax0, cx0), ly = fmaxf(ay0, cy0);
            float rx = fminf(ax1, cx1), ry = fminf(ay1, cy1);
            float w = fmaxf(rx - lx, 0.f), h = fmaxf(ry - ly, 0.f);
            float inter = w * h;
            float iou = inter / (areaA + areaB - inter + 1e-7f);
            if (iou > IOU_THR) supp[t] = 1;
        }
        __syncthreads();
    }
    if (t < KTOP) {
        int fin = (!supp[t]) && vm[t];
        finalf[b * KTOP + t] = fin;
        out[2 * KTOP * 4 + b * KTOP + t] = fin ? ts : 0.0f;
    }
}

// ---------------- Kernel 4: dynamic-conv mask head via WMMA ---------------------
// Per detection: x1 = relu(W1(8x10) @ F(10x6400) + b1); x2 = relu(W2(8x8) @ x1 + b2);
// logit = W3(1x8) @ x2 + b3.   Mapped onto v_wmma_f32_16x16x32_f16:
//   A = weights (16x32, zero-padded), B = 16-pixel feature tile (32x16, zero-padded).
// A lane layout (16-bit A 16x32): lanes 0-15: elems 0..9 = K 0..9 (W1); lanes 16-31 multiply zero B.
// B lane layout (16-bit B 32x16): lane n (0-15): elems 0..15 = K 0..15 of pixel n; lanes 16-31: K16-31 (=0).
// D layout: lane n (0-15) holds M=0..7 (the 8 hidden units of pixel n) in its 8 accumulators.
__global__ void k_maskhead(const float* __restrict__ preds, const float* __restrict__ priors,
                           const float* __restrict__ mf, const int* __restrict__ topidx,
                           float* __restrict__ logits) {
    int det  = blockIdx.x;            // 0..199  (b*100 + k)
    int b    = det / KTOP;
    int lane = threadIdx.x & 31;
    int wave = threadIdx.x >> 5;

    int idx = topidx[det];
    const float* ker = preds + (size_t)(b * NA + idx) * PSTRIDE + (NCLS + 4);
    float prix = priors[idx * 4 + 0];
    float priy = priors[idx * 4 + 1];
    float strd = priors[idx * 4 + 2];
    float inv8s = 1.0f / (strd * 8.0f);

    float b1a[8], b2a[8], w3a[8];
#pragma unroll
    for (int j = 0; j < 8; ++j) {
        w3a[j] = ker[144 + j];
        b1a[j] = ker[152 + j];
        b2a[j] = ker[160 + j];
    }
    float b3 = ker[168];

    // A operands (per-wave constant)
    v16h a1 = {}; v16h a2 = {};
    int m = lane & 15;
    if (lane < 16) {
        if (m < 8) {
#pragma unroll
            for (int i = 0; i < 10; ++i) a1[i] = (_Float16)ker[m * 10 + i];     // W1 K=0..9
#pragma unroll
            for (int i = 0; i < 8; ++i) a2[i] = (_Float16)ker[80 + m * 8 + i]; // W2 K=0..7
        }
    } else {
        if (m < 8) {
            a1[0] = (_Float16)ker[m * 10 + 8];   // W1 K=8
            a1[1] = (_Float16)ker[m * 10 + 9];   // W1 K=9
        }
    }

    v8f zc = {};
    int n = lane & 15;
    for (int tile = wave; tile < HW / 16; tile += 4) {
        int p = tile * 16 + n;
        v16h bv = {};
        if (lane < 16) {
            int pxi = p % 80, pyi = p / 80;
            bv[0] = (_Float16)((prix - (float)(pxi * 8)) * inv8s);
            bv[1] = (_Float16)((priy - (float)(pyi * 8)) * inv8s);
#pragma unroll
            for (int c = 0; c < 8; ++c) bv[2 + c] = (_Float16)mf[c * HW + p];
        }
        v8f d1 = __builtin_amdgcn_wmma_f32_16x16x32_f16(false, a1, false, bv,
                                                        (short)0, zc, false, false);
        v16h b2v = {};
        if (lane < 16) {
#pragma unroll
            for (int j = 0; j < 8; ++j) b2v[j] = (_Float16)fmaxf(d1[j] + b1a[j], 0.0f);
        }
        v8f d2 = __builtin_amdgcn_wmma_f32_16x16x32_f16(false, a2, false, b2v,
                                                        (short)0, zc, false, false);
        if (lane < 16) {
            float acc = b3;
#pragma unroll
            for (int j = 0; j < 8; ++j) acc += w3a[j] * fmaxf(d2[j] + b2a[j], 0.0f);
            logits[(size_t)det * HW + p] = acc;
        }
    }
}

// ---------------- Kernel 5: fused bilinear 80->640 + sigmoid>0.5 + final --------
// sigmoid(v) > 0.5  <=>  v > 0, so no transcendental. float4 stores: 328 MB total,
// the HBM-store roofline of the whole pipeline (~14 us @ 23.3 TB/s).
__global__ void k_upsample(const float* __restrict__ logits, const int* __restrict__ finalf,
                           float* __restrict__ masks) {
    int q  = blockIdx.x * blockDim.x + threadIdx.x;   // 20,480,000 exactly
    int xq = q % 160;
    int y  = (q / 160) % 640;
    int bk = q / (160 * 640);
    const float* L = logits + (size_t)bk * HW;
    float fin = finalf[bk] ? 1.0f : 0.0f;

    float sy = (y + 0.5f) * 0.125f - 0.5f;
    int   y0 = (int)floorf(sy);
    float wy = sy - (float)y0;
    int y0c = clampi(y0, 0, 79), y1c = clampi(y0 + 1, 0, 79);
    __builtin_prefetch(L + y0c * 80, 0, 0);   // -> global_prefetch_b8

    float r[4];
#pragma unroll
    for (int i = 0; i < 4; ++i) {
        int   x  = xq * 4 + i;
        float sx = (x + 0.5f) * 0.125f - 0.5f;
        int   x0 = (int)floorf(sx);
        float wx = sx - (float)x0;
        int x0c = clampi(x0, 0, 79), x1c = clampi(x0 + 1, 0, 79);
        float v00 = L[y0c * 80 + x0c], v01 = L[y0c * 80 + x1c];
        float v10 = L[y1c * 80 + x0c], v11 = L[y1c * 80 + x1c];
        float v = v00 * (1.f - wy) * (1.f - wx) + v01 * (1.f - wy) * wx +
                  v10 * wy * (1.f - wx)         + v11 * wy * wx;
        r[i] = (v > 0.0f) ? fin : 0.0f;
    }
    reinterpret_cast<float4*>(masks)[q] = make_float4(r[0], r[1], r[2], r[3]);
}

// ---------------- Launch ---------------------------------------------------------
extern "C" void kernel_launch(void* const* d_in, const int* in_sizes, int n_in,
                              void* d_out, int out_size, void* d_ws, size_t ws_size,
                              hipStream_t stream) {
    (void)in_sizes; (void)n_in; (void)out_size; (void)ws_size;
    const float* preds  = (const float*)d_in[0];   // (2, 8400, 253)
    const float* mfeat  = (const float*)d_in[1];   // (1, 8, 80, 80)
    const float* priors = (const float*)d_in[2];   // (8400, 4)
    // d_in[3] (coord) reconstructed analytically: coord[p] = ((p%80)*8, (p/80)*8)
    float* out = (float*)d_out;

    char*  ws     = (char*)d_ws;
    float* scores = (float*)(ws + 0);        // 16800 f32
    int*   labels = (int*)  (ws + 67200);    // 16800 i32
    float* tops   = (float*)(ws + 134400);   // 200 f32
    int*   topidx = (int*)  (ws + 135200);   // 200 i32
    int*   finalf = (int*)  (ws + 136000);   // 200 i32
    float* logits = (float*)(ws + 136800);   // 2*100*6400 f32 (5.12 MB)

    k_scores  <<<(2 * NA + 255) / 256, 256, 0, stream>>>(preds, scores, labels);
    k_topk    <<<2, 256, 0, stream>>>(scores, tops, topidx);
    k_nms     <<<2, 128, 0, stream>>>(preds, priors, labels, tops, topidx, out, finalf);
    k_maskhead<<<200, 128, 0, stream>>>(preds, priors, mfeat, topidx, logits);
    k_upsample<<<80000, 256, 0, stream>>>(logits, finalf, out + 1200);
}
